// Quant3Linear_dummy_32762010534269
// MI455X (gfx1250) — compile-verified, matrix operand order checked by reading, not run on hardware
//
#include <hip/hip_runtime.h>
#include <hip/hip_bf16.h>

typedef __attribute__((ext_vector_type(16))) _Float16 v16h;
typedef __attribute__((ext_vector_type(8)))  _Float16 v8h;
typedef __attribute__((ext_vector_type(8)))  float    v8f;

#define IN_F   8192
#define OUT_F  28672
#define GROUPS (IN_F / 32)        // 256 K-chunks of 32 values (3 packed dwords each)
#define KSPLIT 4                  // waves cooperating on one column tile (K split)
#define TILES_PER_BLOCK 2         // 16-column tiles per block
#define GROUPS_PER_WAVE (GROUPS / KSPLIT)   // 64
#define PF_DIST 8                 // prefetch distance in K-chunks

// HBM-bound GEMV: 88 MB of packed weights @ 23.3 TB/s -> ~3.8 us floor.
// To get enough bytes in flight, each 16-column tile is K-split across 4
// waves (7168 waves total), each walking 64 K-chunks.  Per chunk:
//   - 3 coalesced dword loads of the packed qweight rows
//   - global_prefetch_b8 8 chunks ahead (fire-and-forget MLP)
//   - 16 3-bit extracts per lane (v_lshrrev_b64 + and + v_cvt_f16_u16) to
//     build the f16 B fragment in the ISA 7.12.2 register layout
//   - 2 ds_load_b128 of the x_hi / x_lo f16 chunks (A rows 0 and 1)
//   - 1 v_wmma_f32_16x16x32_f16 accumulating into C (f32)
// x is pre-split into f16 hi + f16 residual in LDS, so C[0][n] + C[1][n]
// reproduces the fp32 dot product to ~2^-22 relative.  K-split partials are
// combined through LDS in a fixed order (deterministic).
__global__ __launch_bounds__(256) void q3_gemv_wmma(
    const float* __restrict__ x,
    const int*   __restrict__ qw,
    const float* __restrict__ scales,
    const float* __restrict__ zeros,
    const float* __restrict__ bias,
    float*       __restrict__ out)
{
    __shared__ __align__(16) _Float16 xhi[IN_F];
    __shared__ __align__(16) _Float16 xlo[IN_F];
    __shared__ float sred[256];
    __shared__ float partial[KSPLIT][TILES_PER_BLOCK][16];
    __shared__ float sx_total;

    const int tid = threadIdx.x;

    // Stage x as hi/lo f16 split in LDS and fold sum(x) into the same pass.
    float p = 0.f;
    for (int j = 0; j < IN_F / 256; ++j) {
        int i = tid + j * 256;
        float v = x[i];
        _Float16 h = (_Float16)v;
        xhi[i] = h;
        xlo[i] = (_Float16)(v - (float)h);
        p += v;
    }
    sred[tid] = p;
    __syncthreads();

    // Block-level reduction of sum(x): one wave folds 256 partials.
    if (tid < 32) {
        float s = 0.f;
        #pragma unroll
        for (int j = 0; j < 8; ++j) s += sred[tid + 32 * j];
        #pragma unroll
        for (int m = 16; m >= 1; m >>= 1) s += __shfl_xor(s, m, 32);
        if (tid == 0) sx_total = s;
    }
    __syncthreads();
    const float sx = sx_total;

    const int lane  = tid & 31;
    const int wave  = tid >> 5;
    const int n     = lane & 15;  // column within tile / N index
    const int hi    = lane >> 4;  // lane half selects the K sub-pattern
    const int tileb = wave & (TILES_PER_BLOCK - 1);  // tile within block
    const int ks    = wave >> 1;                     // K-split index 0..3
    const int col   = (blockIdx.x * TILES_PER_BLOCK + tileb) * 16 + n;

    const int g0 = ks * GROUPS_PER_WAVE;
    const int g1 = g0 + GROUPS_PER_WAVE;

    v8f c = {};
    #pragma unroll 2
    for (int g = g0; g < g1; ++g) {
        // Prefetch the packed stream PF_DIST chunks ahead (global_prefetch_b8,
        // no LOADcnt cost).  Clamped index keeps control flow uniform.
        int gp = g + PF_DIST < g1 ? g + PF_DIST : g1 - 1;
        __builtin_prefetch(&qw[(3 * gp + 0) * OUT_F + col], 0, 3);
        __builtin_prefetch(&qw[(3 * gp + 1) * OUT_F + col], 0, 3);
        __builtin_prefetch(&qw[(3 * gp + 2) * OUT_F + col], 0, 3);

        // Three packed words hold 32 3-bit codes for this column, forming a
        // contiguous 96-bit little-endian stream: code j = bits [3j, 3j+3).
        unsigned w0 = (unsigned)qw[(3 * g + 0) * OUT_F + col];
        unsigned w1 = (unsigned)qw[(3 * g + 1) * OUT_F + col];
        unsigned w2 = (unsigned)qw[(3 * g + 2) * OUT_F + col];
        unsigned long long lo64 = ((unsigned long long)w1 << 32) | w0; // bits 0..63
        unsigned long long hi64 = ((unsigned long long)w2 << 32) | w1; // bits 32..95

        // B fragment (32x16 f16): lane = column; VGPR/half -> K per ISA layout.
        v16h b = {};
        #pragma unroll
        for (int r = 0; r < 8; ++r) {
            #pragma unroll
            for (int hlf = 0; hlf < 2; ++hlf) {
                int k  = (r < 4 ? 0 : 16) + (hi ? 8 : 0) + 2 * (r & 3) + hlf;
                int sh = 3 * k;
                unsigned long long src = (k <= 20) ? lo64 : hi64;
                int shv                = (k <= 20) ? sh   : (sh - 32);
                unsigned short code = (unsigned short)((src >> shv) & 7ull);
                b[2 * r + hlf] = (_Float16)code;   // v_cvt_f16_u16
            }
        }

        // A fragment (16x32 f16): row 0 = x_hi chunk, row 1 = x_lo chunk.
        // K pattern: elements 0..7  <-> k = 8*hi + 0..7
        //            elements 8..15 <-> k = 16 + 8*hi + 0..7
        // -> two contiguous 16-byte LDS loads per participating lane.
        v16h a = {};
        if (n < 2) {
            const _Float16* srcx = (n == 0) ? &xhi[g * 32 + hi * 8]
                                            : &xlo[g * 32 + hi * 8];
            v8h lo8 = *(const v8h*)(srcx);
            v8h hi8 = *(const v8h*)(srcx + 16);
            #pragma unroll
            for (int i = 0; i < 8; ++i) { a[i] = lo8[i]; a[8 + i] = hi8[i]; }
        }

        // EXEC is all-1s here (divergence above is closed before the WMMA).
        c = __builtin_amdgcn_wmma_f32_16x16x32_f16(
                /*neg_a=*/false, a, /*neg_b=*/false, b,
                /*c_mod=*/(short)0, c, /*reuse_a=*/false, /*reuse_b=*/false);
    }

    // C layout: VGPR v, lanes 0..15 hold M=v, N=lane.  Lane n (<16) has
    // C[0][n] (x_hi part) in c[0] and C[1][n] (x_lo part) in c[1].
    if (hi == 0) partial[ks][tileb][n] = c[0] + c[1];
    __syncthreads();

    // One wave per tile combines the 4 K-split partials (fixed order) and
    // applies scale/zero/bias.
    if (wave < TILES_PER_BLOCK && hi == 0) {
        int oc = (blockIdx.x * TILES_PER_BLOCK + wave) * 16 + n;
        float Sq = ((partial[0][wave][n] + partial[1][wave][n]) +
                    (partial[2][wave][n] + partial[3][wave][n]));
        out[oc] = scales[oc] * Sq - zeros[oc] * sx + bias[oc];
    }
}

extern "C" void kernel_launch(void* const* d_in, const int* in_sizes, int n_in,
                              void* d_out, int out_size, void* d_ws, size_t ws_size,
                              hipStream_t stream) {
    (void)in_sizes; (void)n_in; (void)out_size; (void)d_ws; (void)ws_size;
    const float* x      = (const float*)d_in[0];
    const int*   qw     = (const int*)  d_in[1];
    const float* scales = (const float*)d_in[2];
    const float* zeros  = (const float*)d_in[3];
    const float* bias   = (const float*)d_in[4];
    float*       out    = (float*)d_out;

    dim3 grid(OUT_F / (16 * TILES_PER_BLOCK));  // 896 blocks
    dim3 block(256);                            // 8 wave32: 2 tiles x 4 K-splits
    q3_gemv_wmma<<<grid, block, 0, stream>>>(x, qw, scales, zeros, bias, out);
}
